// BuNN_10797547782311
// MI455X (gfx1250) — matrix operand order repaired and checked by reading, not compiled, not run on hardware
//
#include <hip/hip_runtime.h>

// ---------------------------------------------------------------------------
// BuNN on MI455X (gfx1250): dominant cost is 32 sweeps of L (16384x16384) in
// term <- (-1/k) * L @ term.  ~400x memory-bound => convert L to bf16 once
// (halves HBM traffic: 32 GiB -> 16 GiB ~ 0.7 ms @ 23.3 TB/s) and drive the
// math with V_WMMA_F32_16X16X32_BF16 (fp32 accumulate).  The GEMM is K-split
// 4-ways per 16-row strip (4096 waves) with manually double-buffered loads so
// waves overlap VMEM with WMMA issue.  Falls back to on-the-fly fp32->bf16
// conversion if the workspace can't hold L_bf (512 MB).
// ---------------------------------------------------------------------------

#define NN   16384   // nodes
#define DIN  128
#define DOUT 40
#define TD   64      // total dim (2*B)
#define NB   32      // bundles
#define HID  64
#define NL   4
#define KT   8       // Taylor steps

typedef __attribute__((ext_vector_type(16))) __bf16 v16bf;
typedef __attribute__((ext_vector_type(8)))  __bf16 v8bf;
typedef __attribute__((ext_vector_type(8)))  float  v8f;

__device__ __forceinline__ v16bf cat16(v8bf lo, v8bf hi) {
    return __builtin_shufflevector(lo, hi, 0,1,2,3,4,5,6,7,8,9,10,11,12,13,14,15);
}

__device__ __forceinline__ float gelu_exact(float x) {
    return 0.5f * x * (1.0f + erff(x * 0.70710678118654752440f));
}

// -------------------------- L fp32 -> bf16 ---------------------------------
__global__ __launch_bounds__(256) void convert_L(const float* __restrict__ L,
                                                 __bf16* __restrict__ Lbf) {
    size_t i = ((size_t)blockIdx.x * blockDim.x + threadIdx.x) * 8;
    v8f f = *reinterpret_cast<const v8f*>(L + i);
    v8bf b;
#pragma unroll
    for (int j = 0; j < 8; ++j) b[j] = (__bf16)f[j];
    *reinterpret_cast<v8bf*>(Lbf + i) = b;
}

// -------------------------- h = x @ emb_W^T + b ----------------------------
__global__ __launch_bounds__(256) void embed_kernel(const float* __restrict__ x,
                                                    const float* __restrict__ W,
                                                    const float* __restrict__ b,
                                                    float* __restrict__ h) {
    int tid = blockIdx.x * blockDim.x + threadIdx.x;   // NN*TD threads
    int col = tid & (TD - 1);
    int row = tid >> 6;
    const float* xr = x + (size_t)row * DIN;
    const float* wr = W + (size_t)col * DIN;
    float s = b[col];
#pragma unroll 8
    for (int j = 0; j < DIN; ++j) s += xr[j] * wr[j];
    h[tid] = s;
}

// --------- per-layer: angles, rotate(transpose=T), H = rot@ltW^T + b -------
// Writes: result = H (f32), termT = bf16(H) transposed (TD x NN), c/s buffers.
__global__ __launch_bounds__(256) void layer_pre(
    const float* __restrict__ h,
    const float* __restrict__ W1, const float* __restrict__ b1,
    const float* __restrict__ W2, const float* __restrict__ b2,
    const float* __restrict__ ltW, const float* __restrict__ ltb,
    float* __restrict__ result, __bf16* __restrict__ termT,
    float* __restrict__ cbuf, float* __restrict__ sbuf) {
    __shared__ float sW1[HID * TD];     // 16 KB
    __shared__ float sW2[NB * HID];     //  8 KB
    __shared__ float sLT[TD * TD];      // 16 KB
    __shared__ float sh[4][TD], sg[4][HID], shr[4][TD];
    __shared__ float sc[4][NB], ss[4][NB];

    int t = threadIdx.x;                // 0..255
    for (int i = t; i < HID * TD; i += 256) sW1[i] = W1[i];
    for (int i = t; i < NB * HID; i += 256) sW2[i] = W2[i];
    for (int i = t; i < TD * TD; i += 256) sLT[i] = ltW[i];

    int r = t >> 6;                     // row slot 0..3
    int c = t & 63;
    int row = blockIdx.x * 4 + r;
    sh[r][c] = h[(size_t)row * TD + c];
    __syncthreads();

    // hidden = gelu(W1 @ h + b1)
    float a1 = b1[c];
#pragma unroll 8
    for (int j = 0; j < TD; ++j) a1 += sW1[c * TD + j] * sh[r][j];
    sg[r][c] = gelu_exact(a1);
    __syncthreads();

    // angles = W2 @ hidden + b2 ; c = cos, s = sin
    if (c < NB) {
        float ang = b2[c];
#pragma unroll 8
        for (int j = 0; j < HID; ++j) ang += sW2[c * HID + j] * sg[r][j];
        float cv = cosf(ang), sv = sinf(ang);
        sc[r][c] = cv; ss[r][c] = sv;
        cbuf[(size_t)row * NB + c] = cv;
        sbuf[(size_t)row * NB + c] = sv;
    }
    __syncthreads();

    // rotate into bundle frame (transpose=True): xr = c*x - s*y ; yr = s*x + c*y
    {
        int bnd = c >> 1;
        float cv = sc[r][bnd], sv = ss[r][bnd];
        float xv = sh[r][bnd * 2], yv = sh[r][bnd * 2 + 1];
        shr[r][c] = ((c & 1) == 0) ? (cv * xv - sv * yv) : (sv * xv + cv * yv);
    }
    __syncthreads();

    // H = ltW @ hr + ltb ; seed Taylor state
    float H = ltb[c];
#pragma unroll 8
    for (int j = 0; j < TD; ++j) H += sLT[c * TD + j] * shr[r][j];
    result[(size_t)row * TD + c] = H;
    termT[(size_t)c * NN + row] = (__bf16)H;
}

// ------------- Taylor step: newterm = coef * (L @ term) --------------------
// Workgroup = 8 waves = 2 M-strips (16 rows x 64 cols) x 4 K-splits.
// Each wave: double-buffered A (L) + B (termT, transposed bf16) fragments,
// 4x v_wmma_f32_16x16x32_bf16 per k-step.  LDS reduction over K-splits, then
// result += coef*sum and bf16 termT_out emission.
template <bool PRECONV>
__global__ __launch_bounds__(256) void taylor_gemm(
    const void* __restrict__ Lsrc,
    const __bf16* __restrict__ termT_in,
    __bf16* __restrict__ termT_out,
    float* __restrict__ result, float coef) {
    __shared__ float red[4 * 2048];      // 32 KB: [ksplit][msub*1024 + m*64 + n]

    const int lane  = threadIdx.x & 31;
    const int wave  = threadIdx.x >> 5;
    const int msub  = wave & 1;          // which 16-row strip
    const int ks    = wave >> 1;         // K-split 0..3
    const int rmod  = lane & 15;
    const int khalf = lane >> 4;
    const int m0    = (blockIdx.x * 2 + msub) * 16;
    const size_t Ns = (size_t)NN;
    const int kbeg  = ks * (NN / 4);
    const int kend  = kbeg + (NN / 4);
    const size_t arowBase = (size_t)(m0 + rmod) * Ns;

    const __bf16* Lbf = (const __bf16*)Lsrc;
    const float*  Lf  = (const float*)Lsrc;

    v8f acc[4] = {v8f{}, v8f{}, v8f{}, v8f{}};

    // Pipeline registers (next fragment in flight while current is consumed)
    v8bf a_lo{}, a_hi{};
    v8f  af_lo{}, af_hi{};
    v8bf b_lo[4], b_hi[4];

    auto loadA = [&](int k0) {
        if constexpr (PRECONV) {
            a_lo = *reinterpret_cast<const v8bf*>(Lbf + arowBase + k0 + khalf * 8);
            a_hi = *reinterpret_cast<const v8bf*>(Lbf + arowBase + k0 + 16 + khalf * 8);
        } else {
            af_lo = *reinterpret_cast<const v8f*>(Lf + arowBase + k0 + khalf * 8);
            af_hi = *reinterpret_cast<const v8f*>(Lf + arowBase + k0 + 16 + khalf * 8);
        }
    };
    auto loadB = [&](int k0) {
#pragma unroll
        for (int nt = 0; nt < 4; ++nt) {
            const __bf16* bp = termT_in + (size_t)(nt * 16 + rmod) * Ns + k0 + khalf * 8;
            b_lo[nt] = *reinterpret_cast<const v8bf*>(bp);
            b_hi[nt] = *reinterpret_cast<const v8bf*>(bp + 16);
        }
    };
    auto makeA = [&]() -> v16bf {
        if constexpr (PRECONV) {
            return cat16(a_lo, a_hi);
        } else {
            v16bf a;
#pragma unroll
            for (int j = 0; j < 8; ++j) { a[j] = (__bf16)af_lo[j]; a[8 + j] = (__bf16)af_hi[j]; }
            return a;
        }
    };

    loadA(kbeg);
    loadB(kbeg);
    for (int k0 = kbeg; k0 < kend - 32; k0 += 32) {
        v16bf a  = makeA();
        v16bf b0 = cat16(b_lo[0], b_hi[0]);
        v16bf b1 = cat16(b_lo[1], b_hi[1]);
        v16bf b2 = cat16(b_lo[2], b_hi[2]);
        v16bf b3 = cat16(b_lo[3], b_hi[3]);
        loadA(k0 + 32);                  // next fragment in flight
        loadB(k0 + 32);
        acc[0] = __builtin_amdgcn_wmma_f32_16x16x32_bf16(false, a, false, b0, (short)0, acc[0], false, false);
        acc[1] = __builtin_amdgcn_wmma_f32_16x16x32_bf16(false, a, false, b1, (short)0, acc[1], false, false);
        acc[2] = __builtin_amdgcn_wmma_f32_16x16x32_bf16(false, a, false, b2, (short)0, acc[2], false, false);
        acc[3] = __builtin_amdgcn_wmma_f32_16x16x32_bf16(false, a, false, b3, (short)0, acc[3], false, false);
    }
    {   // epilogue step (no further loads)
        v16bf a  = makeA();
        v16bf b0 = cat16(b_lo[0], b_hi[0]);
        v16bf b1 = cat16(b_lo[1], b_hi[1]);
        v16bf b2 = cat16(b_lo[2], b_hi[2]);
        v16bf b3 = cat16(b_lo[3], b_hi[3]);
        acc[0] = __builtin_amdgcn_wmma_f32_16x16x32_bf16(false, a, false, b0, (short)0, acc[0], false, false);
        acc[1] = __builtin_amdgcn_wmma_f32_16x16x32_bf16(false, a, false, b1, (short)0, acc[1], false, false);
        acc[2] = __builtin_amdgcn_wmma_f32_16x16x32_bf16(false, a, false, b2, (short)0, acc[2], false, false);
        acc[3] = __builtin_amdgcn_wmma_f32_16x16x32_bf16(false, a, false, b3, (short)0, acc[3], false, false);
    }

    // Stash partials: C layout — VGPR r: M = r + 8*khalf, N = nt*16 + rmod
#pragma unroll
    for (int nt = 0; nt < 4; ++nt) {
#pragma unroll
        for (int r = 0; r < 8; ++r) {
            int m_local = r + 8 * khalf;
            red[ks * 2048 + msub * 1024 + m_local * 64 + nt * 16 + rmod] = acc[nt][r];
        }
    }
    __syncthreads();

    // Reduce over K-splits; coalesced result writes, strided bf16 termT writes.
#pragma unroll
    for (int e = 0; e < 8; ++e) {
        int o = e * 256 + threadIdx.x;           // 0..2047, bank-conflict free
        float v = (red[o] + red[2048 + o] + red[4096 + o] + red[6144 + o]) * coef;
        int n = o & 63;
        int m = blockIdx.x * 32 + (o >> 6);      // msub*16 + m_local folded in
        result[(size_t)m * TD + n] += v;
        termT_out[(size_t)n * Ns + m] = (__bf16)v;
    }
}

// -------- h += gelu(rotate_back(result)) (transpose=False) -----------------
__global__ __launch_bounds__(256) void layer_post(float* __restrict__ h,
                                                  const float* __restrict__ result,
                                                  const float* __restrict__ cbuf,
                                                  const float* __restrict__ sbuf) {
    int tid = blockIdx.x * blockDim.x + threadIdx.x;   // NN*TD
    int c = tid & (TD - 1);
    int row = tid >> 6;
    int bnd = c >> 1;
    float cv = cbuf[(size_t)row * NB + bnd];
    float sv = sbuf[(size_t)row * NB + bnd];
    float xv = result[(size_t)row * TD + bnd * 2];
    float yv = result[(size_t)row * TD + bnd * 2 + 1];
    float v = ((c & 1) == 0) ? (cv * xv + sv * yv) : (-sv * xv + cv * yv);
    h[tid] += gelu_exact(v);
}

// -------------------------- out = h @ out_W^T + b --------------------------
__global__ __launch_bounds__(256) void out_kernel(const float* __restrict__ h,
                                                  const float* __restrict__ W,
                                                  const float* __restrict__ b,
                                                  float* __restrict__ out) {
    int tid = blockIdx.x * blockDim.x + threadIdx.x;
    if (tid >= NN * DOUT) return;
    int row = tid / DOUT;
    int o   = tid - row * DOUT;
    const float* hr = h + (size_t)row * TD;
    const float* wr = W + (size_t)o * TD;
    float s = b[o];
#pragma unroll 8
    for (int j = 0; j < TD; ++j) s += hr[j] * wr[j];
    out[tid] = s;
}

// ---------------------------------------------------------------------------
extern "C" void kernel_launch(void* const* d_in, const int* in_sizes, int n_in,
                              void* d_out, int out_size, void* d_ws, size_t ws_size,
                              hipStream_t stream) {
    const float* x      = (const float*)d_in[0];
    const float* L      = (const float*)d_in[1];
    const float* emb_W  = (const float*)d_in[2];
    const float* emb_b  = (const float*)d_in[3];
    const float* phi_W1 = (const float*)d_in[4];   // (NL, HID, TD)
    const float* phi_b1 = (const float*)d_in[5];   // (NL, HID)
    const float* phi_W2 = (const float*)d_in[6];   // (NL, NB, HID)
    const float* phi_b2 = (const float*)d_in[7];   // (NL, NB)
    const float* lt_W   = (const float*)d_in[8];   // (NL, TD, TD)
    const float* lt_b   = (const float*)d_in[9];   // (NL, TD)
    const float* out_W  = (const float*)d_in[10];  // (DOUT, TD)
    const float* out_b  = (const float*)d_in[11];
    float* out = (float*)d_out;

    // Workspace carve-up
    char* ws = (char*)d_ws;
    float*  result = (float*)ws;  ws += (size_t)NN * TD * 4;
    float*  hbuf   = (float*)ws;  ws += (size_t)NN * TD * 4;
    float*  cbuf   = (float*)ws;  ws += (size_t)NN * NB * 4;
    float*  sbuf   = (float*)ws;  ws += (size_t)NN * NB * 4;
    __bf16* termA  = (__bf16*)ws; ws += (size_t)TD * NN * 2;
    __bf16* termB  = (__bf16*)ws; ws += (size_t)TD * NN * 2;
    size_t baseBytes = (size_t)(ws - (char*)d_ws);
    __bf16* Lbf = (__bf16*)ws;
    const bool preconv = ws_size >= baseBytes + (size_t)NN * NN * 2;

    if (preconv) {
        // 268M elements, 8 per thread
        convert_L<<<(unsigned)((size_t)NN * NN / 8 / 256), 256, 0, stream>>>(L, Lbf);
    }

    embed_kernel<<<NN * TD / 256, 256, 0, stream>>>(x, emb_W, emb_b, hbuf);

    for (int i = 0; i < NL; ++i) {
        layer_pre<<<NN / 4, 256, 0, stream>>>(
            hbuf,
            phi_W1 + (size_t)i * HID * TD, phi_b1 + (size_t)i * HID,
            phi_W2 + (size_t)i * NB * HID, phi_b2 + (size_t)i * NB,
            lt_W + (size_t)i * TD * TD,   lt_b + (size_t)i * TD,
            result, termA, cbuf, sbuf);

        __bf16* tin = termA;
        __bf16* tout = termB;
        for (int k = 1; k <= KT; ++k) {
            float coef = -1.0f / (float)k;   // T = 1.0
            if (preconv)
                taylor_gemm<true><<<NN / 32, 256, 0, stream>>>(Lbf, tin, tout, result, coef);
            else
                taylor_gemm<false><<<NN / 32, 256, 0, stream>>>(L, tin, tout, result, coef);
            __bf16* t = tin; tin = tout; tout = t;
        }

        layer_post<<<NN * TD / 256, 256, 0, stream>>>(hbuf, result, cbuf, sbuf);
    }

    out_kernel<<<(NN * DOUT + 255) / 256, 256, 0, stream>>>(hbuf, out_W, out_b, out);
}
